// CVAE_77498389889529
// MI455X (gfx1250) — matrix operand order repaired and checked by reading, not compiled, not run on hardware
//
#include <hip/hip_runtime.h>
#include <hip/hip_bf16.h>

// ---------------------------------------------------------------------------
// MI455X (gfx1250) CVAE forward. wave32. All heavy GEMMs via
// v_wmma_f32_16x16x32_f16 (f16 inputs, f32 accumulate). Weights pre-cast +
// transposed to N-major f16. GEMM stages the shared 64x64 B panel into LDS
// through the Tensor Data Mover (tensor_load_to_lds + s_wait_tensorcnt),
// with LDS row padding against bank conflicts. Attention is flash-style:
// one wave per (batch, head, 16-query tile), scores and P@V both WMMA.
// ---------------------------------------------------------------------------

typedef __attribute__((ext_vector_type(16))) _Float16 v16h;
typedef __attribute__((ext_vector_type(8)))  float    v8f;

#define WMMA_F16(A, B, C) \
  __builtin_amdgcn_wmma_f32_16x16x32_f16(false, (A), false, (B), (short)0, (C), false, false)

#if defined(__has_builtin)
#if __has_builtin(__builtin_amdgcn_tensor_load_to_lds) && \
    __has_builtin(__builtin_amdgcn_s_wait_tensorcnt)
#define HAS_TDM 1
#endif
#endif
#ifndef HAS_TDM
#define HAS_TDM 0
#endif

__device__ __forceinline__ float act_apply(float v, int act) {
  if (act == 1) v = fmaxf(v, 0.0f);                         // relu
  else if (act == 2) v = tanhf(v);                          // tanh
  else if (act == 3) v = (v > 20.0f) ? v : log1pf(expf(v)); // softplus
  return v;
}

// ------------------------------ PRNG (Box-Muller) ---------------------------
__device__ __forceinline__ unsigned hash_u32(unsigned v) {
  v ^= v >> 16; v *= 0x7feb352du; v ^= v >> 15; v *= 0x846ca68bu; v ^= v >> 16;
  return v;
}
__device__ __forceinline__ float gaussrand(unsigned idx, unsigned seed) {
  unsigned h1 = hash_u32(idx * 2u + 0x9E3779B9u * seed + 1u);
  unsigned h2 = hash_u32(idx * 2u + 1u + 0x85EBCA6Bu * seed + 7u);
  float u1 = ((h1 >> 8) + 0.5f) * (1.0f / 16777216.0f);
  float u2 = ((h2 >> 8) + 0.5f) * (1.0f / 16777216.0f);
  return sqrtf(-2.0f * logf(u1)) * __cosf(6.2831853071795864f * u2);
}

// --------------------------- small utility kernels --------------------------
__global__ void k_zero(float* p, int n) {
  int i = blockIdx.x * blockDim.x + threadIdx.x;
  if (i < n) p[i] = 0.0f;
}

__global__ void k_cast16(const float* __restrict__ a, _Float16* __restrict__ o, long n) {
  long i = (long)blockIdx.x * blockDim.x + threadIdx.x;
  if (i < n) o[i] = (_Float16)a[i];
}

// W is KxN row-major f32; Wt becomes NxK row-major f16 (N-major).
__global__ void k_cast_t16(const float* __restrict__ W, _Float16* __restrict__ Wt,
                           int K, int N) {
  long i = (long)blockIdx.x * blockDim.x + threadIdx.x;
  long total = (long)K * N;
  if (i >= total) return;
  long k = i / N, n = i - k * N;
  Wt[n * (long)K + k] = (_Float16)W[i];
}

// ------------------------- FFT + freq mask + iFFT ---------------------------
__global__ void __launch_bounds__(512)
k_freq_mask(const float* __restrict__ x, const float* __restrict__ mfreq,
            float* __restrict__ mtime) {
  const int N = 4096, NT = 512;
  __shared__ float re[4096];
  __shared__ float im[4096];
  __shared__ float amp[2049];
  __shared__ float redx[512];
  __shared__ int   cnt;
  const int b = blockIdx.x, t = threadIdx.x;
  const float* xr = x + (long)b * N;

  for (int i = t; i < N; i += NT) {
    int r = __brev((unsigned)i) >> 20;           // 12-bit reversal
    re[r] = xr[i]; im[r] = 0.0f;
  }
  __syncthreads();
  for (int len = 2; len <= N; len <<= 1) {       // forward FFT (sign -1)
    int half = len >> 1;
    float ang = -6.283185307179586f / (float)len;
    for (int kidx = t; kidx < N / 2; kidx += NT) {
      int g = kidx / half, j = kidx - g * half;
      int i0 = g * len + j, i1 = i0 + half;
      float wr = __cosf(ang * j), wi = __sinf(ang * j);
      float tr = wr * re[i1] - wi * im[i1];
      float ti = wr * im[i1] + wi * re[i1];
      re[i1] = re[i0] - tr; im[i1] = im[i0] - ti;
      re[i0] += tr;         im[i0] += ti;
    }
    __syncthreads();
  }
  float lmax = 0.0f;
  for (int j = t; j <= 2048; j += NT) {
    float a = sqrtf(re[j] * re[j] + im[j] * im[j]);
    amp[j] = a; lmax = fmaxf(lmax, a);
  }
  redx[t] = lmax; __syncthreads();
  for (int s = 256; s > 0; s >>= 1) { if (t < s) redx[t] = fmaxf(redx[t], redx[t + s]); __syncthreads(); }
  float lo = 0.0f, hi = redx[0] + 1.0f;          // k = int(2049*0.3) = 614
  for (int it = 0; it < 24; ++it) {
    float mid = 0.5f * (lo + hi);
    __syncthreads();
    if (t == 0) cnt = 0;
    __syncthreads();
    int c = 0;
    for (int j = t; j <= 2048; j += NT) c += (amp[j] < mid) ? 1 : 0;
    if (c) atomicAdd(&cnt, c);
    __syncthreads();
    if (cnt <= 614) lo = mid; else hi = mid;
  }
  float thr = 0.5f * (lo + hi);
  float m0 = mfreq[0], m1 = mfreq[1];
  for (int j = t; j <= 2048; j += NT)
    if (amp[j] < thr) { re[j] = m0; im[j] = m1; }
  __syncthreads();
  if (t == 0) { im[0] = 0.0f; im[2048] = 0.0f; }
  __syncthreads();
  for (int j = t + 1; j < 2048; j += NT) { re[N - j] = re[j]; im[N - j] = -im[j]; }
  __syncthreads();
  for (int i = t; i < N; i += NT) {              // in-place bit reversal
    int r = __brev((unsigned)i) >> 20;
    if (i < r) {
      float a = re[i]; re[i] = re[r]; re[r] = a;
      float c = im[i]; im[i] = im[r]; im[r] = c;
    }
  }
  __syncthreads();
  for (int len = 2; len <= N; len <<= 1) {       // inverse FFT (sign +1)
    int half = len >> 1;
    float ang = 6.283185307179586f / (float)len;
    for (int kidx = t; kidx < N / 2; kidx += NT) {
      int g = kidx / half, j = kidx - g * half;
      int i0 = g * len + j, i1 = i0 + half;
      float wr = __cosf(ang * j), wi = __sinf(ang * j);
      float tr = wr * re[i1] - wi * im[i1];
      float ti = wr * im[i1] + wi * re[i1];
      re[i1] = re[i0] - tr; im[i1] = im[i0] - ti;
      re[i0] += tr;         im[i0] += ti;
    }
    __syncthreads();
  }
  float* out = mtime + (long)b * N;
  for (int i = t; i < N; i += NT) out[i] = re[i] * (1.0f / 4096.0f);
}

// -------------------------- patch embed (K=18 tiny) -------------------------
__global__ void __launch_bounds__(512)
k_patch_embed(const float* __restrict__ x, const float* __restrict__ Wl,
              const float* __restrict__ bl, float* __restrict__ fl32,
              _Float16* __restrict__ fl16) {
  const int l = blockIdx.x, b = blockIdx.y, t = threadIdx.x;
  __shared__ float p[16];
  __shared__ float feat[18];
  if (t < 16) {
    int gi = l * 8 + t;
    float v = x[(long)b * 4096 + gi];
    if (gi == 4095) v = 0.0f;                 // x_l[:, -1] = 0
    p[t] = v;
  }
  __syncthreads();
  if (t < 18) {
    int j = (t < 9) ? t : (t - 9);
    bool isim = (t >= 9);
    float s = 0.0f;
    for (int tt = 0; tt < 16; ++tt) {
      float ang = -6.283185307179586f * (float)(j * tt) / 16.0f;
      s += p[tt] * (isim ? __sinf(ang) : __cosf(ang));
    }
    feat[t] = s;
  }
  __syncthreads();
  float s = bl[t];
  #pragma unroll
  for (int k = 0; k < 18; ++k) s += feat[k] * Wl[k * 512 + t];
  s = tanhf(s);
  long idx = ((long)b * 511 + l) * 512 + t;
  fl32[idx] = s;
  fl16[idx] = (_Float16)s;
}

// ------------------ TDM: 2D tile (64 rows x 64 f16) -> LDS ------------------
// D# per CDNA5 ISA ch.8: group0 = {flags, lds_addr, global_addr, type=2},
// group1 = {data_size=2B, pad_enable (16B pad every 32 DWORDs), tensor dims,
// tile 64x64, tensor_dim0_stride=K}. Groups 2/3 zero (2D tensor). This
// toolchain's builtin is the 6-arg form: (g0, g1, g2, g3, g4, cpol).
#define BSTRIDE 72   // 64 payload halfs + 8 pad halfs (16B) per staged row

__device__ __forceinline__ void tdm_load_b_tile(const _Float16* gsrc,
                                                unsigned lds_off, int K) {
#if HAS_TDM
  typedef __attribute__((ext_vector_type(4))) unsigned u32x4;
  typedef __attribute__((ext_vector_type(8))) int i32x8;
  typedef __attribute__((ext_vector_type(4))) int i32x4;
  unsigned long long ga = (unsigned long long)gsrc;
  u32x4 g0;
  g0[0] = 1u;                                          // count=1 (valid user D#)
  g0[1] = lds_off;                                     // lds_addr (bytes)
  g0[2] = (unsigned)(ga & 0xffffffffu);                // global_addr[31:0]
  g0[3] = (unsigned)((ga >> 32) & 0x01ffffffu)         // global_addr[56:32]
        | (2u << 30);                                  // type = 2 ("image")
  unsigned td0 = (unsigned)K;                          // tensor dim0 (elements)
  unsigned td1 = 64u;                                  // tensor dim1 (rows)
  i32x8 g1;
  g1[0] = (int)((1u << 16)                             // data_size = 2 bytes
        | (1u << 20)                                   // pad_enable
        | (4u << 22)                                   // pad_interval: 32 DWORDs
        | (3u << 25));                                 // pad_amount: 4 DWORDs
  g1[1] = (int)((td0 & 0xffffu) << 16);                // tensor_dim0[15:0]
  g1[2] = (int)((td0 >> 16) | ((td1 & 0xffffu) << 16));// dim0[31:16] | dim1[15:0]
  g1[3] = (int)((td1 >> 16) | (64u << 16));            // dim1[31:16] | tile_dim0=64
  g1[4] = (int)64u;                                    // tile_dim1=64, tile_dim2=0
  g1[5] = (int)(unsigned)K;                            // tensor_dim0_stride[31:0]
  g1[6] = 0;                                           // stride[47:32], dim1_stride lo
  g1[7] = 0;
  i32x4 gz4 = {0, 0, 0, 0};
  i32x8 gz8 = {0, 0, 0, 0, 0, 0, 0, 0};
  __builtin_amdgcn_tensor_load_to_lds(g0, g1, gz4, gz4, gz8, 0);
#else
  (void)gsrc; (void)lds_off; (void)K;
#endif
}

// ------------------------------ WMMA GEMM -----------------------------------
// C = act(A[MxK,f16] @ Wt^T + bias + res). Wt is N-major f16 (NxK).
// Block = 4 waves = 128 rows x 64 cols; each wave owns a 32x64 strip
// (2 A frags x 4 B frags = 8 WMMAs per 32-K step). The 64x64 B panel is
// staged in LDS via TDM (all 4 waves share it), double-barrier per 64-K chunk.
// grid = (ceil(M/128), N/64). N%64==0, K%64==0 at every call site.
__global__ void __launch_bounds__(128)
k_gemm_wmma(const _Float16* __restrict__ A, const _Float16* __restrict__ Wt,
            const float* __restrict__ bias, float* __restrict__ C32,
            _Float16* __restrict__ C16, const float* __restrict__ res,
            int M, int N, int K, int act) {
  __shared__ _Float16 bstage[64 * BSTRIDE];
  const int lane = threadIdx.x & 31;
  const int wave = threadIdx.x >> 5;
  const int row0 = blockIdx.x * 128 + wave * 32;
  const int n0   = blockIdx.y * 64;
  const int khalf = (lane >> 4);

  v8f zero = {};
  v8f acc[2][4] = {{zero, zero, zero, zero}, {zero, zero, zero, zero}};

  int r0 = row0 + (lane & 15);      if (r0 >= M) r0 = M - 1;
  int r1 = row0 + 16 + (lane & 15); if (r1 >= M) r1 = M - 1;
  const _Float16* pa0 = A + (long)r0 * K + khalf * 8;
  const _Float16* pa1 = A + (long)r1 * K + khalf * 8;

  for (int kc = 0; kc < K; kc += 64) {
    __syncthreads();                 // previous chunk fully consumed
#if HAS_TDM
    if (wave == 0) {
      tdm_load_b_tile(Wt + (long)n0 * K + kc,
                      (unsigned)(unsigned long long)(void*)bstage, K);
      __builtin_amdgcn_s_wait_tensorcnt(0);
    }
#else
    for (int i = threadIdx.x; i < 64 * 64; i += 128) {
      int nloc = i >> 6, kk = i & 63;
      bstage[nloc * BSTRIDE + kk] = Wt[(long)(n0 + nloc) * K + kc + kk];
    }
#endif
    __syncthreads();                 // B panel visible to all waves
    #pragma unroll
    for (int ks = 0; ks < 2; ++ks) {
      const int k = kc + ks * 32;
      v16h a0, a1;
      #pragma unroll
      for (int e = 0; e < 8; ++e) {
        a0[e] = pa0[k + e]; a0[8 + e] = pa0[k + 16 + e];
        a1[e] = pa1[k + e]; a1[8 + e] = pa1[k + 16 + e];
      }
      if (ks == 0 && kc + 128 < K) {
        __builtin_prefetch(pa0 + kc + 128, 0, 1);   // global_prefetch
        __builtin_prefetch(pa1 + kc + 128, 0, 1);
      }
      #pragma unroll
      for (int j = 0; j < 4; ++j) {
        const _Float16* pb =
            &bstage[(j * 16 + (lane & 15)) * BSTRIDE + ks * 32 + khalf * 8];
        v16h bf;
        #pragma unroll
        for (int e = 0; e < 8; ++e) { bf[e] = pb[e]; bf[8 + e] = pb[16 + e]; }
        acc[0][j] = WMMA_F16(a0, bf, acc[0][j]);
        acc[1][j] = WMMA_F16(a1, bf, acc[1][j]);
      }
    }
  }
  #pragma unroll
  for (int half2 = 0; half2 < 2; ++half2) {
    #pragma unroll
    for (int j = 0; j < 4; ++j) {
      #pragma unroll
      for (int i = 0; i < 8; ++i) {
        int rr = row0 + half2 * 16 + i + khalf * 8;
        int cc = n0 + j * 16 + (lane & 15);
        if (rr >= M) continue;
        float v = acc[half2][j][i];
        if (bias) v += bias[cc];
        long idx = (long)rr * N + cc;
        if (res) v += res[idx];
        v = act_apply(v, act);
        if (C32) C32[idx] = v;
        if (C16) C16[idx] = (_Float16)v;
      }
    }
  }
}

// --------------------------- flash-ish attention ----------------------------
__global__ void __launch_bounds__(32)
k_attention(const _Float16* __restrict__ Qm, const _Float16* __restrict__ Km,
            const _Float16* __restrict__ Vm, _Float16* __restrict__ ctx16) {
  const int L = 511, D = 512;
  const int qb = blockIdx.x, h = blockIdx.y, b = blockIdx.z;
  const int lane = threadIdx.x;
  const int khalf = lane >> 4;
  __shared__ float    S[16][512];
  __shared__ _Float16 P[16][512];
  const long base = (long)b * L * D + h * 64;
  const int row0 = qb * 16;

  int qr = row0 + (lane & 15); if (qr > L - 1) qr = L - 1;
  const _Float16* pq = Qm + base + (long)qr * D + khalf * 8;
  v16h a0, a1;
  #pragma unroll
  for (int e = 0; e < 8; ++e) {
    a0[e] = pq[e];      a0[8 + e] = pq[16 + e];
    a1[e] = pq[32 + e]; a1[8 + e] = pq[48 + e];
  }
  v8f zero = {};
  for (int nt = 0; nt < 32; ++nt) {
    int key0 = nt * 16;
    int kc = key0 + (lane & 15); if (kc > L - 1) kc = L - 1;
    const _Float16* pk = Km + base + (long)kc * D + khalf * 8;
    v16h b0, b1;
    #pragma unroll
    for (int e = 0; e < 8; ++e) { b0[e] = pk[e]; b0[8 + e] = pk[16 + e]; b1[e] = pk[32 + e]; b1[8 + e] = pk[48 + e]; }
    v8f s = zero;
    s = WMMA_F16(a0, b0, s);
    s = WMMA_F16(a1, b1, s);
    #pragma unroll
    for (int i = 0; i < 8; ++i) {
      int rr = i + khalf * 8;
      int cc = key0 + (lane & 15);
      S[rr][cc] = (cc < L) ? s[i] * 0.125f : -1e30f;
    }
  }
  __syncthreads();
  for (int rl = 0; rl < 16; ++rl) {
    float m = -1e30f;
    for (int c = lane; c < 512; c += 32) m = fmaxf(m, S[rl][c]);
    #pragma unroll
    for (int off = 16; off > 0; off >>= 1) m = fmaxf(m, __shfl_xor(m, off, 32));
    float ssum = 0.0f;
    for (int c = lane; c < 512; c += 32) {
      float e = __expf(S[rl][c] - m);
      S[rl][c] = e; ssum += e;
    }
    #pragma unroll
    for (int off = 16; off > 0; off >>= 1) ssum += __shfl_xor(ssum, off, 32);
    float inv = 1.0f / ssum;
    for (int c = lane; c < 512; c += 32) P[rl][c] = (_Float16)(S[rl][c] * inv);
  }
  __syncthreads();
  #pragma unroll
  for (int jt = 0; jt < 4; ++jt) {
    v8f c = zero;
    for (int kt = 0; kt < 16; ++kt) {
      v16h ap, bv;
      int pr = lane & 15;
      #pragma unroll
      for (int e = 0; e < 8; ++e) {
        ap[e]     = P[pr][kt * 32 + khalf * 8 + e];
        ap[8 + e] = P[pr][kt * 32 + 16 + khalf * 8 + e];
      }
      int nd = jt * 16 + (lane & 15);
      int kb = kt * 32 + khalf * 8;
      #pragma unroll
      for (int e = 0; e < 16; ++e) {
        int key = kb + e + (e >= 8 ? 8 : 0); if (key > L - 1) key = L - 1;   // P==0 there
        bv[e] = Vm[base + (long)key * D + nd];
      }
      c = WMMA_F16(ap, bv, c);
    }
    #pragma unroll
    for (int i = 0; i < 8; ++i) {
      int rr = row0 + i + khalf * 8;
      int cc = jt * 16 + (lane & 15);
      if (rr < L) ctx16[(long)b * L * D + (long)rr * D + h * 64 + cc] = (_Float16)c[i];
    }
  }
}

// ------------------------------- LayerNorm ----------------------------------
__global__ void k_ln(const float* __restrict__ in, const float* __restrict__ g,
                     const float* __restrict__ bta, float* __restrict__ out32,
                     _Float16* __restrict__ out16, int N) {
  extern __shared__ float red[];
  const long row = blockIdx.x;
  const int t = threadIdx.x;
  float v = in[row * N + t];
  red[t] = v; __syncthreads();
  for (int s = N >> 1; s > 0; s >>= 1) { if (t < s) red[t] += red[t + s]; __syncthreads(); }
  float mean = red[0] / (float)N; __syncthreads();
  float d = v - mean;
  red[t] = d * d; __syncthreads();
  for (int s = N >> 1; s > 0; s >>= 1) { if (t < s) red[t] += red[t + s]; __syncthreads(); }
  float var = red[0] / (float)N;
  float y = d * rsqrtf(var + 1e-5f);
  if (g) y = y * g[t] + bta[t];
  if (out32) out32[row * N + t] = y;
  if (out16) out16[row * N + t] = (_Float16)y;
}

// out_linear at last position only: grid 64, block 64.
__global__ void k_outlin_last(const float* __restrict__ fl3, const float* __restrict__ W,
                              const float* __restrict__ bias, float* __restrict__ flocal) {
  const int b = blockIdx.x, t = threadIdx.x;
  const float* row = fl3 + ((long)b * 511 + 510) * 512;
  float s = bias[t];
  for (int k = 0; k < 512; ++k) s += row[k] * W[k * 64 + t];
  flocal[b * 64 + t] = tanhf(s);
}

__global__ void k_concat2(const float* __restrict__ a, const float* __restrict__ c,
                          float* __restrict__ o, int na, int nb) {
  const int r = blockIdx.x, t = threadIdx.x;
  if (t < na) o[r * (na + nb) + t] = a[r * na + t];
  else if (t < na + nb) o[r * (na + nb) + t] = c[r * nb + (t - na)];
}

// small dense with optional concat second input: grid(ceil(N/128), M), block 128.
__global__ void k_dense_small(const float* __restrict__ A, int K1,
                              const float* __restrict__ A2, int K2,
                              const float* __restrict__ W, const float* __restrict__ bias,
                              float* __restrict__ out, int N, int act) {
  const int m = blockIdx.y;
  const int n = blockIdx.x * blockDim.x + threadIdx.x;
  if (n >= N) return;
  float s = bias ? bias[n] : 0.0f;
  const float* a = A + (long)m * K1;
  for (int k = 0; k < K1; ++k) s += a[k] * W[(long)k * N + n];
  if (A2) {
    const float* a2 = A2 + (long)m * K2;
    for (int k = 0; k < K2; ++k) s += a2[k] * W[(long)(K1 + k) * N + n];
  }
  out[(long)m * N + n] = act_apply(s, act);
}

__global__ void k_sample_z(const float* __restrict__ mu, const float* __restrict__ var,
                           float* __restrict__ z, int n, unsigned seed) {
  int i = blockIdx.x * blockDim.x + threadIdx.x;
  if (i < n) z[i] = gaussrand((unsigned)i, seed) * sqrtf(1e-7f + var[i]) + mu[i];
}

// rec_x + recon-loss partial sums (acc[0]=sum term, acc[1]=sum y).
__global__ void __launch_bounds__(256)
k_recx_loss(const float* __restrict__ x, const int* __restrict__ y,
            const float* __restrict__ mux, const float* __restrict__ varx,
            float* __restrict__ recx, float* __restrict__ acc) {
  __shared__ float r1[256], r2[256];
  const int t = threadIdx.x;
  const long i = (long)blockIdx.x * 256 + t;
  float term = 0.0f, aterm = 0.0f;
  if (i < 262144) {
    float vx0 = varx[i];
    recx[i] = gaussrand((unsigned)i, 2u) * sqrtf(1e-7f + vx0) + mux[i];
    float vx = vx0 + 1e-8f;
    float alpha = (float)y[i];
    float d = x[i] - mux[i];
    term = alpha * (logf(vx) + d * d / vx);
    aterm = alpha;
  }
  r1[t] = term; r2[t] = aterm; __syncthreads();
  for (int s = 128; s > 0; s >>= 1) {
    if (t < s) { r1[t] += r1[t + s]; r2[t] += r2[t + s]; }
    __syncthreads();
  }
  if (t == 0) { atomicAdd(&acc[0], r1[0]); atomicAdd(&acc[1], r2[0]); }
}

__global__ void k_kld(const float* __restrict__ mu, const float* __restrict__ var,
                      float* __restrict__ acc) {
  __shared__ float r[64];
  const int b = threadIdx.x;
  float s = 0.0f;
  for (int j = 0; j < 16; ++j) {
    float m = mu[b * 16 + j], v = var[b * 16 + j];
    s += -0.5f * (1.0f + v - m * m - __expf(v));
  }
  r[b] = s; __syncthreads();
  for (int st = 32; st > 0; st >>= 1) { if (b < st) r[b] += r[b + st]; __syncthreads(); }
  if (b == 0) acc[2] = r[0];
}

__global__ void k_final(const float* __restrict__ acc, float* __restrict__ loss) {
  if (threadIdx.x == 0) {
    float recon = 0.5f * acc[0] / 262144.0f;
    float beta  = acc[1] / 262144.0f;
    float kld   = acc[2] / 64.0f;
    loss[0] = recon + beta * kld;
  }
}

// ---------------------------------------------------------------------------
extern "C" void kernel_launch(void* const* d_in, const int* in_sizes, int n_in,
                              void* d_out, int out_size, void* d_ws, size_t ws_size,
                              hipStream_t stream) {
  (void)in_sizes; (void)n_in; (void)out_size; (void)ws_size;
  const int B = 64, W = 4096, D = 512, DI = 2048, L = 511, CE = 64;
  const int MT = B * L; // 32704

  // ---- inputs (setup_inputs dict order, params flattened depth-first) ----
  const float* x       = (const float*)d_in[0];
  const int*   y       = (const int*)  d_in[1];
  const float* m_freq  = (const float*)d_in[2];
  const float* eg1_w   = (const float*)d_in[3];   const float* eg1_b  = (const float*)d_in[4];
  const float* eg2_w   = (const float*)d_in[5];   const float* eg2_b  = (const float*)d_in[6];
  const float* eml_w   = (const float*)d_in[7];   const float* eml_b  = (const float*)d_in[8];
  const float* wq_w    = (const float*)d_in[9];   const float* wq_b   = (const float*)d_in[10];
  const float* wk_w    = (const float*)d_in[11];  const float* wk_b   = (const float*)d_in[12];
  const float* wv_w    = (const float*)d_in[13];  const float* wv_b   = (const float*)d_in[14];
  const float* afc_w   = (const float*)d_in[15];  const float* afc_b  = (const float*)d_in[16];
  const float* ln1_g   = (const float*)d_in[17];  const float* ln1_b  = (const float*)d_in[18];
  const float* f1_w    = (const float*)d_in[19];  const float* f1_b   = (const float*)d_in[20];
  const float* f2_w    = (const float*)d_in[21];  const float* f2_b   = (const float*)d_in[22];
  const float* ln2_g   = (const float*)d_in[23];  const float* ln2_b  = (const float*)d_in[24];
  const float* out_w   = (const float*)d_in[25];  const float* out_b  = (const float*)d_in[26];
  const float* enc1_w  = (const float*)d_in[27];  const float* enc1_b = (const float*)d_in[28];
  const float* enc2_w  = (const float*)d_in[29];  const float* enc2_b = (const float*)d_in[30];
  const float* fmu_w   = (const float*)d_in[31];  const float* fmu_b  = (const float*)d_in[32];
  const float* fvar_w  = (const float*)d_in[33];  const float* fvar_b = (const float*)d_in[34];
  const float* din_w   = (const float*)d_in[35];  const float* din_b  = (const float*)d_in[36];
  const float* d1_w    = (const float*)d_in[37];  const float* d1_b   = (const float*)d_in[38];
  const float* d2_w    = (const float*)d_in[39];  const float* d2_b   = (const float*)d_in[40];
  const float* fmx_w   = (const float*)d_in[41];  const float* fmx_b  = (const float*)d_in[42];
  const float* fvx_w   = (const float*)d_in[43];  const float* fvx_b  = (const float*)d_in[44];

  // ---- outputs (tuple concat) ----
  float* outp   = (float*)d_out;
  float* o_mux  = outp;
  float* o_varx = outp + 262144;
  float* o_recx = outp + 524288;
  float* o_mu   = outp + 786432;
  float* o_var  = outp + 787456;
  float* o_cond = outp + 788480;
  float* o_loss = outp + 796672;

  // ---- workspace bump allocator ----
  char* ws = (char*)d_ws;
  size_t off = 0;
  auto alloc = [&](size_t bytes) -> char* {
    char* p = ws + off;
    off += (bytes + 255) & ~(size_t)255;
    return p;
  };
  float*     mtime   = (float*)    alloc((size_t)B * W * 4);
  _Float16*  mtime16 = (_Float16*) alloc((size_t)B * W * 2);
  _Float16*  h256_16 = (_Float16*) alloc((size_t)B * 256 * 2);
  float*     fglobal = (float*)    alloc((size_t)B * CE * 4);
  float*     fl32    = (float*)    alloc((size_t)MT * D * 4);
  _Float16*  fl16    = (_Float16*) alloc((size_t)MT * D * 2);
  _Float16*  q16     = (_Float16*) alloc((size_t)MT * D * 2);
  _Float16*  k16     = (_Float16*) alloc((size_t)MT * D * 2);
  _Float16*  v16     = (_Float16*) alloc((size_t)MT * D * 2);
  _Float16*  ctx16   = (_Float16*) alloc((size_t)MT * D * 2);
  float*     tmp32   = (float*)    alloc((size_t)MT * D * 4);
  float*     fl2_32  = (float*)    alloc((size_t)MT * D * 4);
  _Float16*  fl2_16  = (_Float16*) alloc((size_t)MT * D * 2);
  _Float16*  ffnh16  = (_Float16*) alloc((size_t)MT * DI * 2);
  float*     fl3_32  = (float*)    alloc((size_t)MT * D * 4);
  float*     flocal  = (float*)    alloc((size_t)B * CE * 4);
  float*     condpre = (float*)    alloc((size_t)B * 128 * 4);
  float*     h1      = (float*)    alloc((size_t)B * 100 * 4);
  float*     h2      = (float*)    alloc((size_t)B * 100 * 4);
  float*     zlat    = (float*)    alloc((size_t)B * 16 * 4);
  float*     t1      = (float*)    alloc((size_t)B * 100 * 4);
  float*     t2      = (float*)    alloc((size_t)B * 100 * 4);
  float*     dec32   = (float*)    alloc((size_t)B * W * 4);
  _Float16*  dec16   = (_Float16*) alloc((size_t)B * W * 2);
  float*     acc     = (float*)    alloc(64);
  _Float16*  eg1t    = (_Float16*) alloc((size_t)W * 256 * 2);
  _Float16*  eg2t    = (_Float16*) alloc((size_t)256 * 64 * 2);
  _Float16*  wqt     = (_Float16*) alloc((size_t)D * D * 2);
  _Float16*  wkt     = (_Float16*) alloc((size_t)D * D * 2);
  _Float16*  wvt     = (_Float16*) alloc((size_t)D * D * 2);
  _Float16*  afct    = (_Float16*) alloc((size_t)D * D * 2);
  _Float16*  f1t     = (_Float16*) alloc((size_t)D * DI * 2);
  _Float16*  f2t     = (_Float16*) alloc((size_t)DI * D * 2);
  _Float16*  fmxt    = (_Float16*) alloc((size_t)W * W * 2);
  _Float16*  fvxt    = (_Float16*) alloc((size_t)W * W * 2);

  auto castT = [&](const float* Wm, _Float16* Wt, int K, int N) {
    long total = (long)K * N;
    k_cast_t16<<<(int)((total + 255) / 256), 256, 0, stream>>>(Wm, Wt, K, N);
  };
  auto gemm = [&](const _Float16* Am, const _Float16* Wt, const float* bias,
                  float* C32, _Float16* C16, const float* res,
                  int M, int N, int K, int act) {
    k_gemm_wmma<<<dim3((M + 127) / 128, N / 64), 128, 0, stream>>>(
        Am, Wt, bias, C32, C16, res, M, N, K, act);
  };

  k_zero<<<1, 32, 0, stream>>>(acc, 8);

  // weight conversions (f32 -> N-major f16)
  castT(eg1_w, eg1t, W, 256);
  castT(eg2_w, eg2t, 256, 64);
  castT(wq_w, wqt, D, D);   castT(wk_w, wkt, D, D);
  castT(wv_w, wvt, D, D);   castT(afc_w, afct, D, D);
  castT(f1_w, f1t, D, DI);  castT(f2_w, f2t, DI, D);
  castT(fmx_w, fmxt, W, W); castT(fvx_w, fvxt, W, W);

  // 1) freq mask path -> mtime
  k_freq_mask<<<B, 512, 0, stream>>>(x, m_freq, mtime);
  k_cast16<<<(B * W + 255) / 256, 256, 0, stream>>>(mtime, mtime16, (long)B * W);

  // 2) global embedding: relu(mtime@eg1) then tanh(@eg2) -> f_global
  gemm(mtime16, eg1t, eg1_b, nullptr, h256_16, nullptr, B, 256, W, 1);
  gemm(h256_16, eg2t, eg2_b, fglobal, nullptr, nullptr, B, 64, 256, 2);

  // 3) local patch embedding -> fl
  k_patch_embed<<<dim3(L, B), 512, 0, stream>>>(x, eml_w, eml_b, fl32, fl16);

  // 4) transformer block
  gemm(fl16, wqt, wq_b, nullptr, q16, nullptr, MT, D, D, 0);
  gemm(fl16, wkt, wk_b, nullptr, k16, nullptr, MT, D, D, 0);
  gemm(fl16, wvt, wv_b, nullptr, v16, nullptr, MT, D, D, 0);
  k_attention<<<dim3(32, 8, B), 32, 0, stream>>>(q16, k16, v16, ctx16);
  gemm(ctx16, afct, afc_b, tmp32, nullptr, fl32, MT, D, D, 0);
  k_ln<<<MT, D, D * sizeof(float), stream>>>(tmp32, ln1_g, ln1_b, fl2_32, fl2_16, D);
  gemm(fl2_16, f1t, f1_b, nullptr, ffnh16, nullptr, MT, DI, D, 1);
  gemm(ffnh16, f2t, f2_b, tmp32, nullptr, fl2_32, MT, D, DI, 0);
  k_ln<<<MT, D, D * sizeof(float), stream>>>(tmp32, ln2_g, ln2_b, fl3_32, nullptr, D);
  k_outlin_last<<<B, 64, 0, stream>>>(fl3_32, out_w, out_b, flocal);

  // 5) cond = LN(concat(f_global, f_local)) -> d_out cond slot
  k_concat2<<<B, 128, 0, stream>>>(fglobal, flocal, condpre, 64, 64);
  k_ln<<<B, 128, 128 * sizeof(float), stream>>>(condpre, nullptr, nullptr,
      o_cond, nullptr, 128);

  // 6) VAE encoder
  k_dense_small<<<dim3(1, B), 128, 0, stream>>>(x, W, o_cond, 128, enc1_w, enc1_b, h1, 100, 2);
  k_dense_small<<<dim3(1, B), 128, 0, stream>>>(h1, 100, nullptr, 0, enc2_w, enc2_b, h2, 100, 2);
  k_dense_small<<<dim3(1, B), 128, 0, stream>>>(h2, 100, nullptr, 0, fmu_w, fmu_b, o_mu, 16, 0);
  k_dense_small<<<dim3(1, B), 128, 0, stream>>>(h2, 100, nullptr, 0, fvar_w, fvar_b, o_var, 16, 3);
  k_sample_z<<<4, 256, 0, stream>>>(o_mu, o_var, zlat, B * 16, 1u);

  // 7) VAE decoder
  k_dense_small<<<dim3(1, B), 128, 0, stream>>>(zlat, 16, o_cond, 128, din_w, din_b, t1, 100, 0);
  k_dense_small<<<dim3(1, B), 128, 0, stream>>>(t1, 100, nullptr, 0, d1_w, d1_b, t2, 100, 2);
  k_dense_small<<<dim3(32, B), 128, 0, stream>>>(t2, 100, nullptr, 0, d2_w, d2_b, dec32, W, 0);
  k_cast16<<<(B * W + 255) / 256, 256, 0, stream>>>(dec32, dec16, (long)B * W);
  gemm(dec16, fmxt, fmx_b, o_mux, nullptr, nullptr, B, W, W, 0);
  gemm(dec16, fvxt, fvx_b, o_varx, nullptr, nullptr, B, W, W, 3);

  // 8) rec_x + loss
  k_recx_loss<<<(B * W) / 256, 256, 0, stream>>>(x, y, o_mux, o_varx, o_recx, acc);
  k_kld<<<1, 64, 0, stream>>>(o_mu, o_var, acc);
  k_final<<<1, 32, 0, stream>>>(acc, o_loss);
}
